// MPCPlanner_60370060313202
// MI455X (gfx1250) — matrix operand order repaired and checked by reading, not compiled, not run on hardware
//
#include <hip/hip_runtime.h>
#include <hip/hip_bf16.h>
#include <math.h>

typedef __attribute__((ext_vector_type(16))) _Float16 v16h;
typedef __attribute__((ext_vector_type(8)))  _Float16 v8h;
typedef __attribute__((ext_vector_type(8)))  float    v8f;

namespace {
constexpr int PH = 12;
constexpr int BATCH = 16;
constexpr int H = 256;
constexpr int Z = 64;
constexpr int Z2 = 80;                  // Z + wrb column (64) + zero pad to tile boundary
constexpr int CAND = 512;
constexpr int TOPKN = 64;
constexpr int ITERS = 3;
constexpr int NTOOLS = 36;
constexpr int M_TOT = BATCH * CAND;     // 8192
constexpr int KAUG = H + 32;            // 288 = 9 k-tiles of 32 (cols 256..258 = actions, 259 = bias-1)
constexpr int NKT  = KAUG / 32;         // 9
constexpr int KCH  = KAUG / 8;          // 36 eight-f16 chunks per row
constexpr int LDA = KAUG + 8;           // padded LDS row stride (f16)
constexpr int SST = Z2 + 4;             // 84: padded f32 stride for s tile
constexpr int TILE_M = 64;              // rows per workgroup
constexpr int THREADS = 512;            // 16 waves (wave32)
}

// ---------- fast tanh: native v_tanh_f32 on gfx1250 ----------
__device__ __forceinline__ float fast_tanh(float x) {
#if __has_builtin(__builtin_amdgcn_tanhf)
  return __builtin_amdgcn_tanhf(x);
#else
  return tanhf(x);
#endif
}

// ---------- small deterministic RNG (counter-hash) ----------
__device__ __forceinline__ unsigned mix32(unsigned x) {
  x ^= x >> 16; x *= 0x7feb352dU;
  x ^= x >> 15; x *= 0x846ca68bU;
  x ^= x >> 16; return x;
}
__device__ __forceinline__ float u01(unsigned x) {
  return (float)(x >> 8) * (1.0f / 16777216.0f);
}

// ---------- WMMA fragment helpers (ISA 7.12.2 layouts) ----------
__device__ __forceinline__ v16h load_a_frag(const _Float16* base, int ld, int r0, int k0, int lane) {
  const _Float16* p = base + (size_t)(r0 + (lane & 15)) * ld + k0 + ((lane >> 4) << 3);
  v8h lo = *(const v8h*)(p);
  v8h hi = *(const v8h*)(p + 16);
  return __builtin_shufflevector(lo, hi, 0,1,2,3,4,5,6,7,8,9,10,11,12,13,14,15);
}
__device__ __forceinline__ v16h load_b_frag(const _Float16* base, int ld, int n0, int k0, int lane) {
  const _Float16* p = base + (size_t)(n0 + (lane & 15)) * ld + k0 + ((lane >> 4) << 3);
  v8h lo = *(const v8h*)(p);
  v8h hi = *(const v8h*)(p + 16);
  return __builtin_shufflevector(lo, hi, 0,1,2,3,4,5,6,7,8,9,10,11,12,13,14,15);
}
__device__ __forceinline__ v8f wmma_f16(v16h a, v16h b, v8f c) {
  return __builtin_amdgcn_wmma_f32_16x16x32_f16(false, a, false, b, (short)0, c, false, false);
}

// ---------- kernel 1: transpose+augment weights to f16 ----------
__global__ void prep_weights_kernel(const float* __restrict__ Wb, const float* __restrict__ Wa,
                                    const float* __restrict__ bb, const float* __restrict__ Ws,
                                    const float* __restrict__ bs, const float* __restrict__ wrb,
                                    _Float16* __restrict__ WbT, _Float16* __restrict__ WsT) {
  int i = blockIdx.x * blockDim.x + threadIdx.x;
  if (i < H * KAUG) {
    int n = i / KAUG, k = i % KAUG;
    float v = 0.0f;
    if (k < H)            v = Wb[k * H + n];       // Wb^T
    else if (k < H + 3)   v = Wa[(k - H) * H + n]; // Wa rows folded into K
    else if (k == H + 3)  v = bb[n];               // bias column
    WbT[i] = (_Float16)v;
  }
  if (i < Z2 * KAUG) {
    int n = i / KAUG, k = i % KAUG;
    float v = 0.0f;
    if (n < Z) {
      if (k < H)           v = Ws[k * Z + n];      // Ws^T
      else if (k == H + 3) v = bs[n];
    } else if (n == Z) {
      if (k < H)           v = wrb[k];             // extra output col: b·wrb (no bias, no tanh)
    }
    WsT[i] = (_Float16)v;
  }
}

// ---------- kernel 2: init CEM state ----------
__global__ void init_state_kernel(float* __restrict__ mean, float* __restrict__ stdv,
                                  float* __restrict__ freq) {
  int i = blockIdx.x * blockDim.x + threadIdx.x;
  if (i < PH * BATCH * 2) { mean[i] = 0.0f; stdv[i] = 1.0f; }
  if (i < PH * BATCH * NTOOLS) freq[i] = 1.0f / (float)NTOOLS;
}

// ---------- kernel 3: sample candidate actions ----------
__global__ void sample_actions_kernel(const float* __restrict__ mean, const float* __restrict__ stdv,
                                      const float* __restrict__ freq, float* __restrict__ act, int it) {
  int idx = blockIdx.x * blockDim.x + threadIdx.x;
  if (idx >= PH * M_TOT) return;
  int t = idx / M_TOT, m = idx % M_TOT, b = m / CAND;
  unsigned base = (unsigned)idx * 0x9E3779B9u + (unsigned)it * 0x85EBCA77u;
  float u1 = u01(mix32(base + 0x068bc21fu));
  float u2 = u01(mix32(base + 0x2545f491u));
  float u3 = u01(mix32(base + 0x8f1bbcdcu));
  const float* f = freq + (size_t)(t * BATCH + b) * NTOOLS;
  float cdf = 0.0f; int di = NTOOLS - 1;
  for (int j = 0; j < NTOOLS; ++j) { cdf += f[j]; if (u1 < cdf) { di = j; break; } }
  float r = sqrtf(-2.0f * logf(fmaxf(u2, 1e-7f)));
  float ang = 6.28318530718f * u3;
  float n0 = r * cosf(ang), n1 = r * sinf(ang);
  const float* mp = mean + (size_t)(t * BATCH + b) * 2;
  const float* sp = stdv + (size_t)(t * BATCH + b) * 2;
  float c0 = fminf(fmaxf(mp[0] + sp[0] * n0, -1.0f), 1.0f);
  float c1 = fminf(fmaxf(mp[1] + sp[1] * n1, -1.0f), 1.0f);
  float* ap = act + (size_t)idx * 3;
  ap[0] = (float)di; ap[1] = c0; ap[2] = c1;
}

// ---------- kernel 4: persistent-LDS WMMA rollout ----------
// One workgroup owns 64 rows for all 12 steps. Weights + belief tile resident in LDS (~255 KB);
// each wave additionally caches its 18 Wb B-fragments (144 VGPRs) in registers for all steps.
__global__ __launch_bounds__(THREADS)
void rollout_kernel(const float* __restrict__ belief,
                    const _Float16* __restrict__ gWbT, const _Float16* __restrict__ gWsT,
                    const float* __restrict__ wrs,
                    const float* __restrict__ gact, float* __restrict__ gret) {
  extern __shared__ __align__(16) char smem_raw[];
  _Float16* sWb = (_Float16*)smem_raw;               // [H][LDA]
  _Float16* sWs = sWb + (size_t)H * LDA;             // [Z2][LDA]
  _Float16* sB  = sWs + (size_t)Z2 * LDA;            // [TILE_M][LDA] augmented belief tile
  float* sS   = (float*)(sB + (size_t)TILE_M * LDA); // [TILE_M][SST]: cols 0..63 tanh(s), col 64 = b·wrb
  float* swrs = sS + (size_t)TILE_M * SST;           // [Z]
  float* sret = swrs + Z;                            // [TILE_M]
  float* sPart = sret + TILE_M;                      // [TILE_M][8]

  const int tid  = threadIdx.x;
  const int lane = tid & 31;
  const int wv   = tid >> 5;                         // 0..15
  const int m0   = blockIdx.x * TILE_M;
  const int batch = m0 / CAND;

  __builtin_prefetch(gWbT, 0, 1);

  // stage weights into LDS as 16B chunks
  for (int c = tid; c < H * KCH; c += THREADS) {
    int r = c / KCH, kc = (c % KCH) * 8;
    *(v8h*)(sWb + r * LDA + kc) = *(const v8h*)(gWbT + (size_t)r * KAUG + kc);
  }
  for (int c = tid; c < Z2 * KCH; c += THREADS) {
    int r = c / KCH, kc = (c % KCH) * 8;
    *(v8h*)(sWs + r * LDA + kc) = *(const v8h*)(gWsT + (size_t)r * KAUG + kc);
  }
  for (int i = tid; i < Z; i += THREADS) swrs[i] = wrs[i];
  // init augmented belief tile: b0 = repeat(belief), bias col (k=259) = 1
  for (int c = tid; c < TILE_M * KCH; c += THREADS) {
    int r = c / KCH, kc = (c % KCH) * 8;
    v8h val;
    if (kc < H) {
      const float* bp = belief + (size_t)batch * H + kc;
#pragma unroll
      for (int j = 0; j < 8; ++j) val[j] = (_Float16)bp[j];
    } else {
#pragma unroll
      for (int j = 0; j < 8; ++j) val[j] = (_Float16)0.0f;
      if (kc == H) val[3] = (_Float16)1.0f;          // k = 259 bias column
    }
    *(v8h*)(sB + r * LDA + kc) = val;
  }
  if (tid < TILE_M) sret[tid] = 0.0f;
  __syncthreads();

  // b-GEMM tiling: wave = 32 rows x 32 cols (2x2 WMMA tiles)
  const int rg = (wv & 1) * 32;                      // row base: 0 or 32
  const int cg = (wv >> 1) * 32;                     // col base: 0..224
  // s-GEMM tiling: 20 tiles over 64x80; wave wv takes (wv%4, wv/4); waves 0..3 also (wv, 4)
  const int r0s = (wv & 3) * 16;
  const int n0s = (wv >> 2) * 16;

  // register-resident Wb fragments: 2 col-tiles x 9 k-tiles, live across all 12 steps
  v16h wB[2][NKT];
#pragma unroll
  for (int j = 0; j < 2; ++j)
#pragma unroll
    for (int kt = 0; kt < NKT; ++kt)
      wB[j][kt] = load_b_frag(sWb, LDA, cg + j * 16, kt * 32, lane);

  for (int t = 0; t < PH; ++t) {
    // this step's actions into the augmentation columns
    if (tid < TILE_M * 3) {
      int r = tid / 3, c = tid % 3;
      sB[r * LDA + H + c] = (_Float16)gact[((size_t)t * M_TOT + (m0 + r)) * 3 + c];
    }
    __syncthreads();

    // b_{t+1} = tanh(b_aug @ WbT_aug^T): only A fragments stream from LDS, B is in registers
    v8f acc00 = {}, acc01 = {}, acc10 = {}, acc11 = {};
#pragma unroll
    for (int kt = 0; kt < NKT; ++kt) {
      v16h a0 = load_a_frag(sB, LDA, rg,      kt * 32, lane);
      v16h a1 = load_a_frag(sB, LDA, rg + 16, kt * 32, lane);
      acc00 = wmma_f16(a0, wB[0][kt], acc00);
      acc01 = wmma_f16(a0, wB[1][kt], acc01);
      acc10 = wmma_f16(a1, wB[0][kt], acc10);
      acc11 = wmma_f16(a1, wB[1][kt], acc11);
    }
    __syncthreads();   // all reads of old b done before overwrite

    // native v_tanh + writeback new belief (f16)
    {
      const int colA = cg + (lane & 15), colB = colA + 16;
      const int rb0 = rg + ((lane >> 4) << 3), rb1 = rb0 + 16;
#pragma unroll
      for (int v = 0; v < 8; ++v) {
        sB[(rb0 + v) * LDA + colA] = (_Float16)fast_tanh(acc00[v]);
        sB[(rb0 + v) * LDA + colB] = (_Float16)fast_tanh(acc01[v]);
        sB[(rb1 + v) * LDA + colA] = (_Float16)fast_tanh(acc10[v]);
        sB[(rb1 + v) * LDA + colB] = (_Float16)fast_tanh(acc11[v]);
      }
    }
    __syncthreads();

    // s-GEMM over 64x80 (cols 64 = b·wrb fold): tile (wv%4, wv/4)
    {
      v8f accs = {};
#pragma unroll
      for (int kt = 0; kt < NKT; ++kt) {
        v16h as = load_a_frag(sB, LDA, r0s, kt * 32, lane);
        v16h bs = load_b_frag(sWs, LDA, n0s, kt * 32, lane);
        accs = wmma_f16(as, bs, accs);
      }
      const int col = n0s + (lane & 15);
      const int rb = r0s + ((lane >> 4) << 3);
#pragma unroll
      for (int v = 0; v < 8; ++v) sS[(rb + v) * SST + col] = fast_tanh(accs[v]);
    }
    if (wv < 4) {        // wave-uniform branch: EXEC stays all-ones for WMMA
      v8f accs = {};
#pragma unroll
      for (int kt = 0; kt < NKT; ++kt) {
        v16h as = load_a_frag(sB, LDA, wv * 16, kt * 32, lane);
        v16h bs = load_b_frag(sWs, LDA, 64, kt * 32, lane);
        accs = wmma_f16(as, bs, accs);
      }
      const int col = 64 + (lane & 15);
      const int rb = wv * 16 + ((lane >> 4) << 3);
#pragma unroll
      for (int v = 0; v < 8; ++v) sS[(rb + v) * SST + col] = accs[v];  // raw: b·wrb at col 64
    }
    __syncthreads();

    // returns += (b·wrb from col 64) + tanh(s)·wrs — deterministic 2-phase reduction
    {
      const int row = tid >> 3, seg = tid & 7;
      const float* srow = sS + row * SST + seg * 8;
      const float* wsv = swrs + seg * 8;
      float p = 0.0f;
#pragma unroll
      for (int c = 0; c < 8; ++c) p += srow[c] * wsv[c];
      if (seg == 0) p += sS[row * SST + 64];
      sPart[row * 8 + seg] = p;
    }
    __syncthreads();
    if (tid < TILE_M) {
      float s = 0.0f;
#pragma unroll
      for (int j = 0; j < 8; ++j) s += sPart[tid * 8 + j];
      sret[tid] += s;
    }
    __syncthreads();
  }

  if (tid < TILE_M) gret[m0 + tid] = sret[tid];
}

// ---------- kernel 5: per-batch top-64 + CEM statistics update ----------
__global__ void topk_update_kernel(const float* __restrict__ ret, const float* __restrict__ act,
                                   float* __restrict__ mean, float* __restrict__ stdv,
                                   float* __restrict__ freq) {
  __shared__ float vals[CAND];
  __shared__ float rv[256]; __shared__ int ri[256];
  __shared__ int   sel[TOPKN];
  __shared__ float sa0[TOPKN], sa1[TOPKN]; __shared__ int sdi[TOPKN];
  const int b = blockIdx.x, tid = threadIdx.x;
  vals[tid] = ret[b * CAND + tid];
  vals[tid + 256] = ret[b * CAND + 256 + tid];
  __syncthreads();
  for (int k = 0; k < TOPKN; ++k) {
    float v0 = vals[tid], v1 = vals[tid + 256];
    float mv; int mi;
    if (v1 > v0) { mv = v1; mi = tid + 256; } else { mv = v0; mi = tid; }
    rv[tid] = mv; ri[tid] = mi;
    __syncthreads();
    for (int s = 128; s > 0; s >>= 1) {
      if (tid < s) {
        if (rv[tid + s] > rv[tid] || (rv[tid + s] == rv[tid] && ri[tid + s] < ri[tid])) {
          rv[tid] = rv[tid + s]; ri[tid] = ri[tid + s];
        }
      }
      __syncthreads();
    }
    if (tid == 0) { sel[k] = ri[0]; vals[ri[0]] = -__builtin_inff(); }
    __syncthreads();
  }
  for (int ph = 0; ph < PH; ++ph) {
    if (tid < TOPKN) {
      const float* ap = act + ((size_t)ph * M_TOT + b * CAND + sel[tid]) * 3;
      int di = (int)ap[0];
      sdi[tid] = (di < 0) ? 0 : (di >= NTOOLS ? NTOOLS - 1 : di);
      sa0[tid] = ap[1]; sa1[tid] = ap[2];
    }
    __syncthreads();
    if (tid == 0) {
      float s0 = 0, s1 = 0, q0 = 0, q1 = 0;
      float hist[NTOOLS];
      for (int j = 0; j < NTOOLS; ++j) hist[j] = 1.0f;
      for (int k = 0; k < TOPKN; ++k) {
        s0 += sa0[k]; s1 += sa1[k];
        q0 += sa0[k] * sa0[k]; q1 += sa1[k] * sa1[k];
        hist[sdi[k]] += 1.0f;
      }
      float m0 = s0 / TOPKN, m1 = s1 / TOPKN;
      float v0 = fmaxf(q0 / TOPKN - m0 * m0, 0.0f);
      float v1 = fmaxf(q1 / TOPKN - m1 * m1, 0.0f);
      size_t base = (size_t)(ph * BATCH + b);
      mean[base * 2 + 0] = m0; mean[base * 2 + 1] = m1;
      stdv[base * 2 + 0] = sqrtf(v0); stdv[base * 2 + 1] = sqrtf(v1);
      for (int j = 0; j < NTOOLS; ++j) freq[base * NTOOLS + j] = hist[j] * 0.01f; // /(64+36)
    }
    __syncthreads();
  }
}

// ---------- kernel 6: finalize output ----------
__global__ void finalize_kernel(const float* __restrict__ freq, const float* __restrict__ mean,
                                float* __restrict__ out) {
  int b = threadIdx.x;
  if (b >= BATCH) return;
  const float* f = freq + (size_t)b * NTOOLS;   // ph = 0
  int arg = 0; float best = f[0];
  for (int j = 1; j < NTOOLS; ++j) if (f[j] > best) { best = f[j]; arg = j; }
  out[b * 3 + 0] = (float)arg;
  out[b * 3 + 1] = mean[b * 2 + 0];
  out[b * 3 + 2] = mean[b * 2 + 1];
}

extern "C" void kernel_launch(void* const* d_in, const int* in_sizes, int n_in,
                              void* d_out, int out_size, void* d_ws, size_t ws_size,
                              hipStream_t stream) {
  (void)in_sizes; (void)n_in; (void)out_size; (void)ws_size;
  const float* belief = (const float*)d_in[0];
  const float* Wb  = (const float*)d_in[2];
  const float* Wa  = (const float*)d_in[3];
  const float* bb  = (const float*)d_in[4];
  const float* Ws  = (const float*)d_in[5];
  const float* bs  = (const float*)d_in[6];
  const float* wrb = (const float*)d_in[7];
  const float* wrs = (const float*)d_in[8];
  float* out = (float*)d_out;

  char* ws = (char*)d_ws;
  size_t off = 0;
  auto take = [&](size_t bytes) { char* p = ws + off; off = (off + bytes + 255) & ~(size_t)255; return p; };
  _Float16* WbT = (_Float16*)take((size_t)H * KAUG * sizeof(_Float16));
  _Float16* WsT = (_Float16*)take((size_t)Z2 * KAUG * sizeof(_Float16));
  float* act  = (float*)take((size_t)PH * M_TOT * 3 * sizeof(float));
  float* gret = (float*)take((size_t)M_TOT * sizeof(float));
  float* mean = (float*)take((size_t)PH * BATCH * 2 * sizeof(float));
  float* stdv = (float*)take((size_t)PH * BATCH * 2 * sizeof(float));
  float* freq = (float*)take((size_t)PH * BATCH * NTOOLS * sizeof(float));

  constexpr size_t SMEM = (size_t)(H + Z2 + TILE_M) * LDA * sizeof(_Float16)
                        + (size_t)TILE_M * SST * sizeof(float)
                        + (size_t)(Z + TILE_M) * sizeof(float)
                        + (size_t)TILE_M * 8 * sizeof(float);   // 260,864 B < 320 KB/WGP

  prep_weights_kernel<<<(H * KAUG + 255) / 256, 256, 0, stream>>>(Wb, Wa, bb, Ws, bs, wrb, WbT, WsT);
  init_state_kernel<<<(PH * BATCH * NTOOLS + 255) / 256, 256, 0, stream>>>(mean, stdv, freq);

  for (int it = 0; it < ITERS; ++it) {
    sample_actions_kernel<<<(PH * M_TOT + 255) / 256, 256, 0, stream>>>(mean, stdv, freq, act, it);
    rollout_kernel<<<M_TOT / TILE_M, THREADS, SMEM, stream>>>(belief, WbT, WsT, wrs, act, gret);
    topk_update_kernel<<<BATCH, 256, 0, stream>>>(gret, act, mean, stdv, freq);
  }
  finalize_kernel<<<1, 32, 0, stream>>>(freq, mean, out);
}
